// AdditiveAttention_48120813585016
// MI455X (gfx1250) — matrix operand order repaired and checked by reading, not compiled, not run on hardware
//
#include <hip/hip_runtime.h>
#include <hip/hip_bf16.h>

// Problem constants (match reference)
constexpr int B  = 4;
constexpr int NQ = 128;
constexpr int NK = 1024;
constexpr int H  = 256;
constexpr int DV = 256;

// LDS row stride for 256-float panels: +4 pad keeps 16B alignment for async
// B128 copies AND spreads the 16 tk-lanes across banks ((h + 4*tk) % 64 distinct).
constexpr int LPAD = 260;

typedef __attribute__((ext_vector_type(16))) _Float16 v16h;
typedef __attribute__((ext_vector_type(8)))  float    v8f;

__device__ __forceinline__ float fast_tanh(float x) {
#if __has_builtin(__builtin_amdgcn_tanhf)
    return __builtin_amdgcn_tanhf(x);   // gfx1250 v_tanh_f32
#else
    return tanhf(x);
#endif
}

#if __has_builtin(__builtin_amdgcn_global_load_async_to_lds_b128) && \
    __has_builtin(__builtin_amdgcn_s_wait_asynccnt)
#define AA_HAVE_ASYNC_LDS 1
// Builtin signature (from hipcc diagnostic): (v4i addrspace(1)*, v4i addrspace(3)*, Imm offset, Imm cpol)
typedef int v4i_t __attribute__((ext_vector_type(4)));
typedef __attribute__((address_space(1))) v4i_t* aa_gptr_t;
typedef __attribute__((address_space(3))) v4i_t* aa_lptr_t;
__device__ __forceinline__ void aa_async_copy16(const float* g, float* l) {
    __builtin_amdgcn_global_load_async_to_lds_b128(
        (aa_gptr_t)const_cast<float*>(g), (aa_lptr_t)l, 0, 0);
}
#else
#define AA_HAVE_ASYNC_LDS 0
#endif

// ---------------- WMMA fragment helpers (wave32, 16x16x32 f16, ISA 7.12.2) ---

// A: 16x32 (MxK) f16. a[e] holds K = khalf*16 + e, M = lane&15.
__device__ __forceinline__ v16h load_frag_a_f32(const float* __restrict__ X,
                                                int ld, int row0, int k0, int lane) {
    int m = lane & 15;
    int khalf = lane >> 4;
    const float* row = X + (size_t)(row0 + m) * ld + k0 + khalf * 16;
    v16h a;
#pragma unroll
    for (int e = 0; e < 8; ++e)  a[e] = (_Float16)row[e];
#pragma unroll
    for (int e = 8; e < 16; ++e) a[e] = (_Float16)row[e];
    return a;
}

__device__ __forceinline__ v16h load_frag_a_f16(const _Float16* __restrict__ X,
                                                int ld, int row0, int k0, int lane) {
    int m = lane & 15;
    int khalf = lane >> 4;
    const _Float16* row = X + (size_t)(row0 + m) * ld + k0 + khalf * 16;
    v16h a;
#pragma unroll
    for (int e = 0; e < 8; ++e)  a[e] = row[e];
#pragma unroll
    for (int e = 8; e < 16; ++e) a[e] = row[e];
    return a;
}

// B: 32x16 (KxN) f16. b[e] = Bmat[khalf*16 + e][lane&15].
// Projection variant: B[k][n] = W[n0+n][k0+k] -> each lane reads 16 contiguous floats.
__device__ __forceinline__ v16h load_frag_b_WT(const float* __restrict__ W,
                                               int ld, int k0, int n0, int lane) {
    int n = lane & 15;
    int khalf = lane >> 4;
    const float* src = W + (size_t)(n0 + n) * ld + k0 + khalf * 16;
    v16h b;
#pragma unroll
    for (int e = 0; e < 16; ++e) b[e] = (_Float16)src[e];
    return b;
}

// Row-major KxN, f32 source.
__device__ __forceinline__ v16h load_frag_b_rm_f32(const float* __restrict__ V,
                                                   int ld, int k0, int n0, int lane) {
    int n = lane & 15;
    int khalf = lane >> 4;
    const float* src = V + (size_t)(k0 + khalf * 16) * ld + n0 + n;
    v16h b;
#pragma unroll
    for (int e = 0; e < 16; ++e) b[e] = (_Float16)src[(size_t)e * ld];
    return b;
}

// Row-major KxN, f16 source (pre-converted values).
__device__ __forceinline__ v16h load_frag_b_rm_f16(const _Float16* __restrict__ V,
                                                   int ld, int k0, int n0, int lane) {
    int n = lane & 15;
    int khalf = lane >> 4;
    const _Float16* src = V + (size_t)(k0 + khalf * 16) * ld + n0 + n;
    v16h b;
#pragma unroll
    for (int e = 0; e < 16; ++e) b[e] = src[(size_t)e * ld];
    return b;
}

// C/D: 16x16 f32, 8 VGPRs. vgpr r: M = r + (lane>=16 ? 8 : 0), N = lane&15.
__device__ __forceinline__ void store_frag_c(float* __restrict__ O, int ld,
                                             int row0, int n0, int lane, v8f c) {
    int n  = lane & 15;
    int mh = (lane >> 4) * 8;
#pragma unroll
    for (int r = 0; r < 8; ++r)
        O[(size_t)(row0 + mh + r) * ld + n0 + n] = c[r];
}

// ---------------- Kernel 1: row-wise projection  P = X @ W^T  (f32 in/out) ---
__global__ void aa_proj_kernel(const float* __restrict__ X, const float* __restrict__ W,
                               float* __restrict__ P) {
    int wave = (blockIdx.x * blockDim.x + threadIdx.x) >> 5;
    int lane = threadIdx.x & 31;
    const int ntileN = H / 16;                 // 16
    int tm = wave / ntileN;
    int tn = wave % ntileN;

    v8f c = {};
#pragma unroll 2
    for (int k0 = 0; k0 < H; k0 += 32) {
        v16h a = load_frag_a_f32(X, H, tm * 16, k0, lane);
        v16h b = load_frag_b_WT(W, H, k0, tn * 16, lane);
        c = __builtin_amdgcn_wmma_f32_16x16x32_f16(false, a, false, b,
                                                   (short)0, c, false, false);
    }
    store_frag_c(P, H, tm * 16, tn * 16, lane, c);
}

// ---------------- Kernel 2: additive scores + mask --------------------------
// s[b,q,k] = sum_h wv[h] * tanh(qp[b,q,h] + kp[b,k,h]);  k>=valid -> -1e6.
// Block = 256 threads = 16x16 (q,k) tile; panels staged in LDS via async copy.
__global__ void aa_score_kernel(const float* __restrict__ qp,
                                const float* __restrict__ kp,
                                const float* __restrict__ wv,
                                const int*   __restrict__ valid_lens,
                                float* __restrict__ scores) {
    const int b  = blockIdx.z;
    const int q0 = blockIdx.y * 16;
    const int k0 = blockIdx.x * 16;

    __shared__ __align__(16) float qs[16 * LPAD];
    __shared__ __align__(16) float ks[16 * LPAD];
    __shared__ float ws[H];

    const int tid = threadIdx.x;
    const float* qbase = qp + ((size_t)b * NQ + q0) * H;
    const float* kbase = kp + ((size_t)b * NK + k0) * H;

#if AA_HAVE_ASYNC_LDS
    // 16 rows x 1024B per panel = 1024 16B-chunks per panel; 256 threads x 4 each.
#pragma unroll
    for (int j = 0; j < 4; ++j) {
        int c   = tid + j * 256;          // chunk id within panel
        int r   = c >> 6;                 // row (64 chunks of 16B per row)
        int c16 = c & 63;                 // 16B chunk within row
        aa_async_copy16(qbase + (size_t)r * H + c16 * 4, &qs[r * LPAD + c16 * 4]);
        aa_async_copy16(kbase + (size_t)r * H + c16 * 4, &ks[r * LPAD + c16 * 4]);
    }
    ws[tid] = wv[tid];
    __builtin_amdgcn_s_wait_asynccnt(0);
    __syncthreads();
#else
    for (int i = tid; i < 16 * H; i += 256) {
        int r = i >> 8;           // H == 256
        int c = i & (H - 1);
        qs[r * LPAD + c] = qbase[(size_t)r * H + c];
        ks[r * LPAD + c] = kbase[(size_t)r * H + c];
    }
    ws[tid] = wv[tid];
    __syncthreads();
#endif

    const int tk = tid & 15;
    const int tq = tid >> 4;
    const float* qr = qs + tq * LPAD;
    const float* kr = ks + tk * LPAD;

    float acc = 0.f;
#pragma unroll 4
    for (int h = 0; h < H; ++h)
        acc = __builtin_fmaf(ws[h], fast_tanh(qr[h] + kr[h]), acc);

    int k = k0 + tk;
    if (k >= valid_lens[b]) acc = -1e6f;
    scores[((size_t)b * NQ + q0 + tq) * NK + k] = acc;
}

// ---------------- Kernel 3: row softmax over NK, emit f16 attn --------------
__global__ void aa_softmax_kernel(const float* __restrict__ scores,
                                  _Float16* __restrict__ attn16) {
    const int row = blockIdx.x;                 // b*NQ + q
    const int tid = threadIdx.x;
    const float* s = scores + (size_t)row * NK;
    __shared__ float red[256];

    float v[4];
    float m = -3.4e38f;
#pragma unroll
    for (int i = 0; i < 4; ++i) { v[i] = s[tid + i * 256]; m = fmaxf(m, v[i]); }
    red[tid] = m;
    __syncthreads();
    for (int off = 128; off > 0; off >>= 1) {
        if (tid < off) red[tid] = fmaxf(red[tid], red[tid + off]);
        __syncthreads();
    }
    m = red[0];
    __syncthreads();

    float sum = 0.f;
#pragma unroll
    for (int i = 0; i < 4; ++i) { v[i] = __expf(v[i] - m); sum += v[i]; }
    red[tid] = sum;
    __syncthreads();
    for (int off = 128; off > 0; off >>= 1) {
        if (tid < off) red[tid] += red[tid + off];
        __syncthreads();
    }
    float inv = 1.0f / red[0];

    _Float16* o = attn16 + (size_t)row * NK;
#pragma unroll
    for (int i = 0; i < 4; ++i) o[tid + i * 256] = (_Float16)(v[i] * inv);
}

// ---------------- values f32 -> f16 pre-convert -----------------------------
__global__ void aa_cvt_values_kernel(const float* __restrict__ V,
                                     _Float16* __restrict__ Vh) {
    size_t i = ((size_t)blockIdx.x * blockDim.x + threadIdx.x) * 4;
#pragma unroll
    for (int j = 0; j < 4; ++j) Vh[i + j] = (_Float16)V[i + j];
}

// ---------------- Kernel 4: out = attn @ values  (WMMA f16 -> f32) ----------
__global__ void aa_pv_kernel_h(const _Float16* __restrict__ attn,
                               const _Float16* __restrict__ valuesh,
                               float* __restrict__ out) {
    int wave = (blockIdx.x * blockDim.x + threadIdx.x) >> 5;
    int lane = threadIdx.x & 31;
    const int TN = DV / 16;                    // 16
    const int TM = NQ / 16;                    // 8
    int b  = wave / (TM * TN);
    int r  = wave % (TM * TN);
    int tm = r / TN;
    int tn = r % TN;

    const _Float16* A = attn    + (size_t)b * NQ * NK;
    const _Float16* V = valuesh + (size_t)b * NK * DV;
    float*          O = out     + (size_t)b * NQ * DV;

    v8f c = {};
    for (int k0 = 0; k0 < NK; k0 += 32) {
        v16h a  = load_frag_a_f16(A, NK, tm * 16, k0, lane);
        v16h bf = load_frag_b_rm_f16(V, DV, k0, tn * 16, lane);
        c = __builtin_amdgcn_wmma_f32_16x16x32_f16(false, a, false, bf,
                                                   (short)0, c, false, false);
    }
    store_frag_c(O, DV, tm * 16, tn * 16, lane, c);
}

// Fallback if workspace can't hold the f16 copy of values.
__global__ void aa_pv_kernel_f(const _Float16* __restrict__ attn,
                               const float* __restrict__ values,
                               float* __restrict__ out) {
    int wave = (blockIdx.x * blockDim.x + threadIdx.x) >> 5;
    int lane = threadIdx.x & 31;
    const int TN = DV / 16;
    const int TM = NQ / 16;
    int b  = wave / (TM * TN);
    int r  = wave % (TM * TN);
    int tm = r / TN;
    int tn = r % TN;

    const _Float16* A = attn   + (size_t)b * NQ * NK;
    const float*    V = values + (size_t)b * NK * DV;
    float*          O = out    + (size_t)b * NQ * DV;

    v8f c = {};
    for (int k0 = 0; k0 < NK; k0 += 32) {
        v16h a  = load_frag_a_f16(A, NK, tm * 16, k0, lane);
        v16h bf = load_frag_b_rm_f32(V, DV, k0, tn * 16, lane);
        c = __builtin_amdgcn_wmma_f32_16x16x32_f16(false, a, false, bf,
                                                   (short)0, c, false, false);
    }
    store_frag_c(O, DV, tm * 16, tn * 16, lane, c);
}

// ---------------- launcher ---------------------------------------------------
extern "C" void kernel_launch(void* const* d_in, const int* in_sizes, int n_in,
                              void* d_out, int out_size, void* d_ws, size_t ws_size,
                              hipStream_t stream) {
    const float* queries    = (const float*)d_in[0];
    const float* keys       = (const float*)d_in[1];
    const float* values     = (const float*)d_in[2];
    const int*   valid_lens = (const int*)  d_in[3];
    const float* Wq         = (const float*)d_in[4];
    const float* Wk         = (const float*)d_in[5];
    const float* wv         = (const float*)d_in[6];
    float*       out        = (float*)d_out;

    // workspace layout
    float*    qp     = (float*)d_ws;                              // B*NQ*H  f32
    float*    kp     = qp + (size_t)B * NQ * H;                   // B*NK*H  f32
    float*    scores = kp + (size_t)B * NK * H;                   // B*NQ*NK f32
    _Float16* attn16 = (_Float16*)(scores + (size_t)B * NQ * NK); // B*NQ*NK f16
    _Float16* vh     = attn16 + (size_t)B * NQ * NK;              // B*NK*DV f16

    size_t need_base = (size_t)(B*NQ*H + B*NK*H + B*NQ*NK) * 4 + (size_t)B*NQ*NK * 2;
    size_t need_vh   = need_base + (size_t)B * NK * DV * 2;
    bool use_vh = (ws_size >= need_vh);

    // 1) projections (WMMA): one wave per 16x16 tile
    {
        int waves = (B * NQ / 16) * (H / 16);                // 512
        aa_proj_kernel<<<waves * 32 / 256, 256, 0, stream>>>(queries, Wq, qp);
    }
    {
        int waves = (B * NK / 16) * (H / 16);                // 4096
        aa_proj_kernel<<<waves * 32 / 256, 256, 0, stream>>>(keys, Wk, kp);
    }

    // 2) additive scores + mask (v_tanh_f32, async-LDS staged)
    {
        dim3 grid(NK / 16, NQ / 16, B);                      // 64 x 8 x 4
        aa_score_kernel<<<grid, 256, 0, stream>>>(qp, kp, wv, valid_lens, scores);
    }

    // 3) softmax over keys, emit f16 attention
    aa_softmax_kernel<<<B * NQ, 256, 0, stream>>>(scores, attn16);

    // 4) attn @ values (WMMA)
    int waves = B * (NQ / 16) * (DV / 16);                   // 512
    if (use_vh) {
        size_t n = (size_t)B * NK * DV;                      // 1,048,576
        aa_cvt_values_kernel<<<(int)(n / 4 / 256), 256, 0, stream>>>(values, vh);
        aa_pv_kernel_h<<<waves * 32 / 256, 256, 0, stream>>>(attn16, vh, out);
    } else {
        aa_pv_kernel_f<<<waves * 32 / 256, 256, 0, stream>>>(attn16, values, out);
    }
}
